// VectorQuantizer_4303557230884
// MI455X (gfx1250) — matrix operand order repaired and checked by reading, not compiled, not run on hardware
//
#include <hip/hip_runtime.h>
#include <hip/hip_bf16.h>
#include <math.h>

typedef __attribute__((ext_vector_type(16))) _Float16 v16h;
typedef __attribute__((ext_vector_type(8)))  float    v8f;
typedef __attribute__((ext_vector_type(4)))  unsigned int v4u;
typedef __attribute__((ext_vector_type(8)))  int      v8i;
typedef __attribute__((ext_vector_type(4)))  int      v4i;

#define NE        1024      // codebook entries
#define EDIM      64        // embedding dim
#define NROWS     65536     // B*L
#define ROWS_BLK  128       // rows per block
#define NBLOCKS   (NROWS / ROWS_BLK)   // 512
#define NTILES    (NE / 16)            // 64 code tiles
#define ZQ_OFF    1
#define ZQ_LEN    (NROWS * EDIM)       // 4194304
#define PERP_OFF  (1 + ZQ_LEN)         // 4194305
#define IDX_OFF   (2 + ZQ_LEN)         // 4194306

// Prep buffer (global, also mirrored at LDS offset 0):
//   [0, 131072)        f16 codebook in WMMA B-layout
//   [131072, 135168)   float ||e||^2 per code
#define PREP_BYTES 135168
#define PREP_QW    (PREP_BYTES / 8)    // 16896 = 0x4200 (8-byte units for TDM)

// Dynamic LDS layout (bytes)
#define SM_BMAT   0                    // _Float16[4096*16]  = 131072
#define SM_ENORM  131072               // float[1024]        = 4096
#define SM_ZST    135168               // _Float16[128*64]   = 16384
#define SM_IDX    151552               // int[128]           = 512
#define SM_LSUM   152064               // float              = 4
#define SM_BYTES  152080

union AB32 { v16h h; uint4 u[2]; };

// ---- one-time prep: codebook -> f16 B-layout + ||e||^2, in global ws ----
// segment s (0..4095): n = s%16, half = (s/16)&1, kk = (s/32)&1, j = s/64
// holds cb[j*16+n][kk*32 + half*16 + 0..15] as 16 contiguous halves.
__global__ __launch_bounds__(256)
void vq_prep_kernel(const float* __restrict__ cb, unsigned short* __restrict__ prep) {
    int s = blockIdx.x * 256 + threadIdx.x;        // 4096 threads
    {
        int n = s & 15, half = (s >> 4) & 1, kk = (s >> 5) & 1, j = s >> 6;
        int e = j * 16 + n;
        int kstart = kk * 32 + half * 16;
        const float4* src = (const float4*)(cb + e * EDIM + kstart);
        _Float16* dst = (_Float16*)prep + s * 16;
        #pragma unroll
        for (int q = 0; q < 4; ++q) {
            float4 v = src[q];
            dst[q * 4 + 0] = (_Float16)v.x;
            dst[q * 4 + 1] = (_Float16)v.y;
            dst[q * 4 + 2] = (_Float16)v.z;
            dst[q * 4 + 3] = (_Float16)v.w;
        }
    }
    if (s < NE) {
        const float4* r4 = (const float4*)(cb + s * EDIM);
        float sum = 0.0f;
        #pragma unroll
        for (int q = 0; q < EDIM / 4; ++q) {
            float4 v = r4[q];
            sum += v.x * v.x + v.y * v.y + v.z * v.z + v.w * v.w;
        }
        ((float*)((char*)prep + SM_ENORM))[s] = sum;
    }
}

__global__ __launch_bounds__(256)
void vq_main_kernel(const float* __restrict__ z, const float* __restrict__ cb,
                    float* __restrict__ out, float* __restrict__ counts,
                    float* __restrict__ sumsq,
                    const unsigned short* __restrict__ prep) {
    extern __shared__ char smem[];
    _Float16* bmat  = (_Float16*)(smem + SM_BMAT);
    float*    enorm = (float*)   (smem + SM_ENORM);
    _Float16* zst   = (_Float16*)(smem + SM_ZST);
    int*      idxsh = (int*)     (smem + SM_IDX);
    float*    lsum  = (float*)   (smem + SM_LSUM);

    const int tid = threadIdx.x;
    const int R0  = blockIdx.x * ROWS_BLK;

    if (tid == 0) *lsum = 0.0f;

    if (prep != nullptr) {
        // ---- TDM: DMA 132KB (B-layout codebook + ||e||^2) global -> LDS ----
        if (tid < 32) {  // wave 0 issues one tensor op for the block
            unsigned long long ga = (unsigned long long)(uintptr_t)prep;
            unsigned int lds0 = (unsigned int)(uintptr_t)smem;   // LDS byte offset
            // D# group0: count=1, lds_addr, global_addr[56:0], type=2
            v4u g0 = { 1u, lds0, (unsigned int)(ga & 0xFFFFFFFFull),
                       (unsigned int)((ga >> 32) & 0x01FFFFFFull) | (2u << 30) };
            // D# group1: data_size=8B; tensor_dim0=tile_dim0=stride0=PREP_QW;
            //            tensor_dim1=tile_dim1=1 (simple 1-row 2D copy)
            v8i g1 = { 0x00030000,                 // data_size=3 (8B)
                       (int)(PREP_QW << 16),       // tensor_dim0 lo16 in [63:48]
                       0x00010000,                 // tensor_dim1=1 in [95:80]
                       (int)(PREP_QW << 16),       // tile_dim0 in [127:112]
                       1,                          // tile_dim1=1 in [143:128]
                       PREP_QW,                    // tensor_dim0_stride lo32
                       0, 0 };
            v4i z4 = { 0, 0, 0, 0 };
            v8i z8 = { 0, 0, 0, 0, 0, 0, 0, 0 };
            __builtin_amdgcn_tensor_load_to_lds(g0, g1, z4, z4, z8, 0);
            __builtin_amdgcn_s_wait_tensorcnt(0);
        }
    } else {
        // ---- fallback: per-block conversion (small ws) ----
        for (int e = tid; e < NE; e += 256) {
            const float4* r4 = (const float4*)(cb + e * EDIM);
            float sum = 0.0f;
            #pragma unroll
            for (int q = 0; q < EDIM / 4; ++q) {
                float4 v = r4[q];
                sum += v.x * v.x + v.y * v.y + v.z * v.z + v.w * v.w;
            }
            enorm[e] = sum;
        }
        for (int s = tid; s < NTILES * 64; s += 256) {
            int n = s & 15, half = (s >> 4) & 1, kk = (s >> 5) & 1, j = s >> 6;
            int e = j * 16 + n;
            int kstart = kk * 32 + half * 16;
            const float4* src = (const float4*)(cb + e * EDIM + kstart);
            _Float16* dst = bmat + s * 16;
            #pragma unroll
            for (int q = 0; q < 4; ++q) {
                float4 v = src[q];
                dst[q * 4 + 0] = (_Float16)v.x;
                dst[q * 4 + 1] = (_Float16)v.y;
                dst[q * 4 + 2] = (_Float16)v.z;
                dst[q * 4 + 3] = (_Float16)v.w;
            }
        }
    }

    // ---- stage block's z rows as f16 (A-layout-friendly row-major) ----
    for (int i4 = tid; i4 < ROWS_BLK * EDIM / 4; i4 += 256) {
        int row = i4 >> 4, c4 = (i4 & 15) * 4;
        float4 v = *(const float4*)(z + (size_t)(R0 + row) * EDIM + c4);
        _Float16* d = zst + row * EDIM + c4;
        d[0] = (_Float16)v.x; d[1] = (_Float16)v.y;
        d[2] = (_Float16)v.z; d[3] = (_Float16)v.w;
    }
    __syncthreads();

    // ---- WMMA distance + fused argmin (double-buffered B tiles) ----
    {
        const int wave = tid >> 5;
        const int lane = tid & 31;
        const int hi   = lane >> 4;
        const int n0   = lane & 15;
        const int hi8  = hi * 8;

        // A-tiles (16x32 f16, contiguous k order: lane half h holds k = 16h..16h+15)
        const int arow = wave * 16 + n0;               // A: M = lane&15
        const uint4* zr = (const uint4*)(zst + arow * EDIM);
        AB32 a0, a1;
        a0.u[0] = zr[(2 * hi8) >> 3];
        a0.u[1] = zr[(8 + 2 * hi8) >> 3];
        a1.u[0] = zr[(32 + 2 * hi8) >> 3];
        a1.u[1] = zr[(40 + 2 * hi8) >> 3];

        float minv[8];
        int   mini[8];
        #pragma unroll
        for (int r = 0; r < 8; ++r) { minv[r] = 3.4e38f; mini[r] = 0; }

        const char* bbase = (const char*)bmat + (hi * 16 + n0) * 32;
        AB32 b0c, b1c, b0n, b1n;
        {   // prologue: tile 0
            const uint4* p0 = (const uint4*)(bbase);
            const uint4* p1 = (const uint4*)(bbase + 1024);
            b0c.u[0] = p0[0]; b0c.u[1] = p0[1];
            b1c.u[0] = p1[0]; b1c.u[1] = p1[1];
        }
        float en_c = enorm[n0];

        for (int j = 0; j < NTILES; ++j) {
            float en_n = 0.0f;
            if (j + 1 < NTILES) {   // prefetch next B tile while WMMAs run
                const uint4* p0 = (const uint4*)(bbase + (j + 1) * 2048);
                const uint4* p1 = (const uint4*)(bbase + (j + 1) * 2048 + 1024);
                b0n.u[0] = p0[0]; b0n.u[1] = p0[1];
                b1n.u[0] = p1[0]; b1n.u[1] = p1[1];
                en_n = enorm[(j + 1) * 16 + n0];
            }
            v8f c = {};
            c = __builtin_amdgcn_wmma_f32_16x16x32_f16(false, a0.h, false, b0c.h,
                                                       (short)0, c, false, false);
            c = __builtin_amdgcn_wmma_f32_16x16x32_f16(false, a1.h, false, b1c.h,
                                                       (short)0, c, false, false);
            // argmin over 0.5*||e||^2 - z.e  (||z||^2 constant per row: dropped)
            float he = 0.5f * en_c;
            int cand = j * 16 + n0;
            #pragma unroll
            for (int r = 0; r < 8; ++r) {
                float d = he - c[r];
                if (d < minv[r]) { minv[r] = d; mini[r] = cand; }
            }
            b0c = b0n; b1c = b1n; en_c = en_n;
        }

        // cross-lane argmin within each 16-lane half (wave32 shuffles)
        #pragma unroll
        for (int m = 8; m >= 1; m >>= 1) {
            #pragma unroll
            for (int r = 0; r < 8; ++r) {
                float ov = __shfl_xor(minv[r], m, 32);
                int   oi = __shfl_xor(mini[r], m, 32);
                if (ov < minv[r] || (ov == minv[r] && oi < mini[r])) {
                    minv[r] = ov; mini[r] = oi;
                }
            }
        }
        if (n0 == 0) {
            #pragma unroll
            for (int r = 0; r < 8; ++r) {
                int rl = wave * 16 + r + hi8;
                idxsh[rl] = mini[r];
                out[IDX_OFF + R0 + rl] = (float)mini[r];
            }
        }
    }
    __syncthreads();

    // ---- gather z_q, loss partial, histogram ----
    float acc = 0.0f;
    for (int i4 = tid; i4 < ROWS_BLK * EDIM / 4; i4 += 256) {
        int row = i4 >> 4, q = i4 & 15;
        int e = idxsh[row];
        float4 cv = *(const float4*)(cb + e * EDIM + q * 4);
        float4 zv = *(const float4*)(z + (size_t)(R0 + row) * EDIM + q * 4);
        float dx = cv.x - zv.x, dy = cv.y - zv.y, dz = cv.z - zv.z, dw = cv.w - zv.w;
        acc += dx * dx + dy * dy + dz * dz + dw * dw;
        float* o = out + ZQ_OFF + (size_t)(R0 + row) * EDIM + q * 4;
        o[0] = cv.x; o[1] = cv.y; o[2] = cv.z; o[3] = cv.w;
    }
    #pragma unroll
    for (int m = 16; m >= 1; m >>= 1) acc += __shfl_xor(acc, m, 32);
    if ((tid & 31) == 0) atomicAdd(lsum, acc);
    if (tid < ROWS_BLK) atomicAdd(&counts[idxsh[tid]], 1.0f);
    __syncthreads();
    if (tid == 0) atomicAdd(sumsq, *lsum);
}

__global__ __launch_bounds__(1024)
void vq_finalize_kernel(const float* __restrict__ counts,
                        const float* __restrict__ sumsq,
                        float* __restrict__ out) {
    __shared__ float warp_h[32];
    int tid = threadIdx.x;
    float em = counts[tid] * (1.0f / (float)NROWS);
    float h = em * logf(em + 1e-10f);
    #pragma unroll
    for (int m = 16; m >= 1; m >>= 1) h += __shfl_xor(h, m, 32);
    if ((tid & 31) == 0) warp_h[tid >> 5] = h;
    __syncthreads();
    if (tid < 32) {
        float v = warp_h[tid];
        #pragma unroll
        for (int m = 16; m >= 1; m >>= 1) v += __shfl_xor(v, m, 32);
        if (tid == 0) {
            out[0]        = 1.25f * (*sumsq) / (float)ZQ_LEN;  // (1+BETA)*MSE
            out[PERP_OFF] = expf(-v);
        }
    }
}

extern "C" void kernel_launch(void* const* d_in, const int* in_sizes, int n_in,
                              void* d_out, int out_size, void* d_ws, size_t ws_size,
                              hipStream_t stream) {
    const float* z  = (const float*)d_in[0];
    const float* cb = (const float*)d_in[1];
    float* out = (float*)d_out;

    (void)in_sizes; (void)n_in; (void)out_size;

    const size_t need = PREP_BYTES + (NE + 1) * sizeof(float);
    const bool use_prep = (ws_size >= need);

    unsigned short* prep = use_prep ? (unsigned short*)d_ws : nullptr;
    float* counts = use_prep ? (float*)((char*)d_ws + PREP_BYTES) : (float*)d_ws;
    float* sumsq  = counts + NE;

    (void)hipFuncSetAttribute((const void*)vq_main_kernel,
                              hipFuncAttributeMaxDynamicSharedMemorySize, SM_BYTES);

    (void)hipMemsetAsync(counts, 0, (NE + 1) * sizeof(float), stream);
    if (use_prep)
        vq_prep_kernel<<<16, 256, 0, stream>>>(cb, prep);
    vq_main_kernel<<<NBLOCKS, 256, SM_BYTES, stream>>>(z, cb, out, counts, sumsq, prep);
    vq_finalize_kernel<<<1, 1024, 0, stream>>>(counts, sumsq, out);
}
